// GNNModel_76656576299278
// MI455X (gfx1250) — matrix operand order repaired and checked by reading, not compiled, not run on hardware
//
#include <hip/hip_runtime.h>
#include <hip/hip_bf16.h>

typedef _Float16 v16h __attribute__((ext_vector_type(16)));
typedef _Float16 v8h  __attribute__((ext_vector_type(8)));
typedef float    v8f  __attribute__((ext_vector_type(8)));

#define NNODES 50000
#define FEAT   128
#define HID    256
#define NGRAPH 512
#define NTASK  4

// ---------------------------------------------------------------- utilities
__global__ __launch_bounds__(256) void fill0_kernel(float* __restrict__ p, size_t n) {
    size_t i = (size_t)blockIdx.x * blockDim.x + threadIdx.x;
    size_t stride = (size_t)gridDim.x * blockDim.x;
    for (; i < n; i += stride) p[i] = 0.0f;
}

__global__ __launch_bounds__(256) void deg_kernel(const int* __restrict__ dst,
                                                  float* __restrict__ deg, int E) {
    int e = blockIdx.x * blockDim.x + threadIdx.x;
    if (e < E) atomicAdd(&deg[dst[e]], 1.0f);
}

// dinv[n] = rsqrt(deg[n] + 1)   (+1 = self loop; always > 0)
__global__ __launch_bounds__(256) void dinv_kernel(float* __restrict__ dinv, int n) {
    int i = blockIdx.x * blockDim.x + threadIdx.x;
    if (i < n) dinv[i] = rsqrtf(dinv[i] + 1.0f);
}

// ---------------------------------------------------------------- WMMA GEMM
// C[M x 256] = A[M x K] * W[K x 256], f32 in/out, f16 WMMA with f32 accum.
// Block = 256 thr (8 waves) computes a 64-row x 256-col output tile.
//   wave w: row-tile rt = w&3, col-tiles (w>>2)*8 .. +7  -> 8 accumulators,
//   one A fragment feeds 8 v_wmma_f32_16x16x32_f16 per K-step.
// All 8 B fragments are preloaded into registers before the WMMA chain so
// ds_load latency overlaps matrix-pipe execution (staggered s_wait_dscnt).
// W is staged per 64-wide K-slice into LDS, transposed + f16-converted:
//   lwt[n][k] with padded stride 72 halves (144B, 16B-aligned rows).
// A-fragment lane layout (matches B below):
//   lane<16: K = kk+{0..15}; lane>=16: K = kk+16+{0..15}.
// B-fragment lane layout (contiguous k order, same as A): lane holds col=lane%16,
//   lanes 0-15: K = kk+{0..15}; lanes 16-31: K = kk+16+{0..15}
//   -> 16 consecutive halves = two aligned ds_load_b128 per fragment.
template<int K>
__global__ __launch_bounds__(256)
void gemm_wmma_f16(const float* __restrict__ A, const float* __restrict__ W,
                   float* __restrict__ C, int M) {
    constexpr int Nout = 256;
    constexpr int KC = 64;            // K-slice staged in LDS
    constexpr int LSTRIDE = KC + 8;   // 72 halves = 144 B (16B-aligned rows)
    __shared__ _Float16 lwt[Nout * LSTRIDE];   // 36,864 B

    const int t    = threadIdx.x;
    const int wave = t >> 5;
    const int lane = t & 31;
    const int l16  = lane & 15;
    const bool hi  = lane >= 16;
    const int rt     = wave & 3;          // row tile within block (0..3)
    const int ctbase = (wave >> 2) * 8;   // first col tile (0 or 8)

    const int row_g = blockIdx.x * 64 + rt * 16 + l16;
    const int row   = row_g < M ? row_g : (M - 1);   // clamp (EXEC stays all-1s)
    const int kbA   = hi ? 16 : 0;

    v8f acc[8];
    #pragma unroll
    for (int c = 0; c < 8; ++c) acc[c] = (v8f){0.f,0.f,0.f,0.f,0.f,0.f,0.f,0.f};

    for (int c0 = 0; c0 < K; c0 += KC) {
        // ---- stage W[c0..c0+63][0..255] into LDS, transposed, f32 -> f16
        #pragma unroll 4
        for (int i = t; i < KC * Nout; i += 256) {
            int k = i >> 8;        // 0..63
            int n = i & 255;       // coalesced over n
            lwt[n * LSTRIDE + k] = (_Float16)W[(size_t)(c0 + k) * Nout + n];
        }
        __syncthreads();

        #pragma unroll
        for (int kk = 0; kk < KC; kk += 32) {
            // ---- A fragment (global, f32 -> f16 in-register)
            const float* ap = A + (size_t)row * K + (c0 + kk + kbA);
            const float4 a0 = *(const float4*)(ap);
            const float4 a1 = *(const float4*)(ap + 4);
            const float4 a2 = *(const float4*)(ap + 8);
            const float4 a3 = *(const float4*)(ap + 12);
            v16h a;
            a[0]=(_Float16)a0.x; a[1]=(_Float16)a0.y; a[2]=(_Float16)a0.z; a[3]=(_Float16)a0.w;
            a[4]=(_Float16)a1.x; a[5]=(_Float16)a1.y; a[6]=(_Float16)a1.z; a[7]=(_Float16)a1.w;
            a[8]=(_Float16)a2.x; a[9]=(_Float16)a2.y; a[10]=(_Float16)a2.z; a[11]=(_Float16)a2.w;
            a[12]=(_Float16)a3.x; a[13]=(_Float16)a3.y; a[14]=(_Float16)a3.z; a[15]=(_Float16)a3.w;

            // ---- preload all 8 B fragments (one ds clause, staggered waits)
            const int k0 = kk + (hi ? 16 : 0);
            v16h bfrag[8];
            #pragma unroll
            for (int c = 0; c < 8; ++c) {
                const int col = (ctbase + c) * 16 + l16;
                const v8h* bp = (const v8h*)&lwt[col * LSTRIDE + k0];
                const v8h b0 = bp[0];
                const v8h b1 = bp[1];
                bfrag[c] = __builtin_shufflevector(b0, b1,
                    0,1,2,3,4,5,6,7,8,9,10,11,12,13,14,15);
            }
            // ---- back-to-back WMMA chain on the matrix pipe
            #pragma unroll
            for (int c = 0; c < 8; ++c) {
                acc[c] = __builtin_amdgcn_wmma_f32_16x16x32_f16(
                    false, a, false, bfrag[c], (short)0, acc[c], false, false);
            }
        }
        __syncthreads();
    }

    // ---- store: VGPR r -> M = r + (lane<16 ? 0 : 8), N = lane%16
    const int mbase = blockIdx.x * 64 + rt * 16 + (hi ? 8 : 0);
    #pragma unroll
    for (int c = 0; c < 8; ++c) {
        const int col = (ctbase + c) * 16 + l16;
        #pragma unroll
        for (int r = 0; r < 8; ++r) {
            const int m = mbase + r;
            if (m < M) C[(size_t)m * Nout + col] = acc[c][r];
        }
    }
}

// ---------------------------------------------------------------- scatter
// agg[dst] += h[src] * dinv[src]*dinv[dst]; 64 lanes per edge, 4 feats/lane.
__global__ __launch_bounds__(256)
void scatter_kernel(const float* __restrict__ h, const int* __restrict__ src,
                    const int* __restrict__ dst, const float* __restrict__ dinv,
                    float* __restrict__ agg, int E) {
    size_t idx = (size_t)blockIdx.x * blockDim.x + threadIdx.x;
    size_t total = (size_t)E * (HID / 4);
    if (idx >= total) return;
    int e = (int)(idx / (HID / 4));
    int f = (int)(idx % (HID / 4)) * 4;
    int s = src[e], d = dst[e];
    float w = dinv[s] * dinv[d];
    const float4 hv = *reinterpret_cast<const float4*>(h + (size_t)s * HID + f);
    float* ap = agg + (size_t)d * HID + f;
    atomicAdd(ap + 0, hv.x * w);
    atomicAdd(ap + 1, hv.y * w);
    atomicAdd(ap + 2, hv.z * w);
    atomicAdd(ap + 3, hv.w * w);
}

// agg[n,f] = relu(agg[n,f] + h[n,f]*dinv[n]^2 + bias[f])   (in place)
__global__ __launch_bounds__(256)
void self_bias_relu_kernel(float* __restrict__ agg, const float* __restrict__ h,
                           const float* __restrict__ dinv, const float* __restrict__ bias,
                           int n) {
    size_t idx = (size_t)blockIdx.x * blockDim.x + threadIdx.x;
    if (idx >= (size_t)n * HID) return;
    int node = (int)(idx / HID);
    int f    = (int)(idx % HID);
    float di = dinv[node];
    float v  = agg[idx] + h[idx] * di * di + bias[f];
    agg[idx] = v > 0.0f ? v : 0.0f;
}

// ---------------------------------------------------------------- pooling
__global__ __launch_bounds__(256) void cnt_kernel(const int* __restrict__ batch,
                                                  float* __restrict__ cnt, int n) {
    int i = blockIdx.x * blockDim.x + threadIdx.x;
    if (i < n) atomicAdd(&cnt[batch[i]], 1.0f);
}

__global__ __launch_bounds__(256)
void pool_sum_kernel(const float* __restrict__ x, const int* __restrict__ batch,
                     float* __restrict__ pool, int n) {
    size_t idx = (size_t)blockIdx.x * blockDim.x + threadIdx.x;
    if (idx >= (size_t)n * HID) return;
    int node = (int)(idx / HID);
    int f    = (int)(idx % HID);
    atomicAdd(&pool[(size_t)batch[node] * HID + f], x[idx]);
}

// ---------------------------------------------------------------- head MLP
__global__ __launch_bounds__(256)
void mlp_kernel(const float* __restrict__ pool, const float* __restrict__ cnt,
                const float* __restrict__ num_atoms,
                const float* __restrict__ W3, const float* __restrict__ b3,
                const float* __restrict__ W4, const float* __restrict__ b4,
                float* __restrict__ out) {
    __shared__ float z[HID + 1];
    __shared__ float hid[32];
    int g = blockIdx.x;
    int t = threadIdx.x;
    float inv = 1.0f / fmaxf(cnt[g], 1.0f);
    if (t < HID) z[t] = pool[(size_t)g * HID + t] * inv;
    if (t == 0) z[HID] = num_atoms[g];
    __syncthreads();
    if (t < 32) {
        float acc = b3[t];
        for (int k = 0; k < HID + 1; ++k) acc += z[k] * W3[k * 32 + t];
        hid[t] = acc > 0.0f ? acc : 0.0f;
    }
    __syncthreads();
    if (t < NTASK) {
        float acc = b4[t];
        for (int k = 0; k < 32; ++k) acc += hid[k] * W4[k * NTASK + t];
        out[(size_t)g * NTASK + t] = acc;
    }
}

// ---------------------------------------------------------------- launcher
extern "C" void kernel_launch(void* const* d_in, const int* in_sizes, int n_in,
                              void* d_out, int out_size, void* d_ws, size_t ws_size,
                              hipStream_t stream) {
    const float* x          = (const float*)d_in[0];
    const int*   edge_index = (const int*)d_in[1];      // [2,E] row-major
    const int*   batch      = (const int*)d_in[2];
    const float* num_atoms  = (const float*)d_in[3];
    const float* W1 = (const float*)d_in[4];
    const float* b1 = (const float*)d_in[5];
    const float* W2 = (const float*)d_in[6];
    const float* b2 = (const float*)d_in[7];
    const float* W3 = (const float*)d_in[8];
    const float* b3 = (const float*)d_in[9];
    const float* W4 = (const float*)d_in[10];
    const float* b4 = (const float*)d_in[11];
    float* out = (float*)d_out;

    const int E = in_sizes[1] / 2;
    const int N = NNODES;
    const int* src = edge_index;       // row 0
    const int* dst = edge_index + E;   // row 1

    // workspace layout (floats): 2x N*HID + N + G*HID + G  (~103 MB)
    float* ws   = (float*)d_ws;
    const size_t nh = (size_t)N * HID;
    float* buf0 = ws;                            // GEMM output h
    float* buf1 = ws + nh;                       // agg / activation
    float* dinv = ws + 2 * nh;                   // N
    float* pool = dinv + N;                      // G*HID
    float* cnt  = pool + (size_t)NGRAPH * HID;   // G

    const int B = 256;
    const int nhBlocks   = (int)((nh + B - 1) / B);
    const int gemmBlocks = (N + 63) / 64;

    // 1) symmetric normalization  dinv = rsqrt(in_deg + 1)
    fill0_kernel<<<(N + B - 1) / B, B, 0, stream>>>(dinv, (size_t)N);
    deg_kernel<<<(E + B - 1) / B, B, 0, stream>>>(dst, dinv, E);
    dinv_kernel<<<(N + B - 1) / B, B, 0, stream>>>(dinv, N);

    // 2) layer 1: h1 = x @ W1  (K=128)
    gemm_wmma_f16<FEAT><<<gemmBlocks, B, 0, stream>>>(x, W1, buf0, N);
    fill0_kernel<<<nhBlocks, B, 0, stream>>>(buf1, nh);
    {
        size_t tot = (size_t)E * (HID / 4);
        scatter_kernel<<<(unsigned)((tot + B - 1) / B), B, 0, stream>>>(buf0, src, dst, dinv, buf1, E);
    }
    self_bias_relu_kernel<<<nhBlocks, B, 0, stream>>>(buf1, buf0, dinv, b1, N);  // buf1 = x2

    // 3) layer 2: h2 = x2 @ W2  (K=256)
    gemm_wmma_f16<HID><<<gemmBlocks, B, 0, stream>>>(buf1, W2, buf0, N);
    fill0_kernel<<<nhBlocks, B, 0, stream>>>(buf1, nh);
    {
        size_t tot = (size_t)E * (HID / 4);
        scatter_kernel<<<(unsigned)((tot + B - 1) / B), B, 0, stream>>>(buf0, src, dst, dinv, buf1, E);
    }
    self_bias_relu_kernel<<<nhBlocks, B, 0, stream>>>(buf1, buf0, dinv, b2, N);  // buf1 = x3

    // 4) global mean pool
    fill0_kernel<<<(int)(((size_t)NGRAPH * HID + B - 1) / B), B, 0, stream>>>(pool, (size_t)NGRAPH * HID);
    fill0_kernel<<<(NGRAPH + B - 1) / B, B, 0, stream>>>(cnt, (size_t)NGRAPH);
    cnt_kernel<<<(N + B - 1) / B, B, 0, stream>>>(batch, cnt, N);
    pool_sum_kernel<<<nhBlocks, B, 0, stream>>>(buf1, batch, pool, N);

    // 5) head MLP
    mlp_kernel<<<NGRAPH, B, 0, stream>>>(pool, cnt, num_atoms, W3, b3, W4, b4, out);
}
